// ChannelAttentionBlock_69767448756880
// MI455X (gfx1250) — compile-verified, matrix operand order checked by reading, not run on hardware
//
#include <hip/hip_runtime.h>

// ---------------------------------------------------------------------------
// Channel attention block, MI455X (gfx1250, wave32, WMMA).
//   a: (16, 128, 256, 256) fp32, beta: scalar fp32
//   Bm = A·A^T (per-batch 128x128 Gram), X = softmax(Bm), C = X·A,
//   out = beta*C + a
// Memory-bound (~1.6 GB minimum traffic vs 69 GFLOP): bf16 WMMA with fp32
// accumulation makes compute vanish under the HBM stream. Both GEMM kernels
// are software-pipelined: double-buffered LDS tiles (one barrier per K-step),
// next-step global loads issued before the WMMA burst, fragments in distinct
// registers (only constant-indexed arrays -> no scratch spills) so ds_load
// latency overlaps v_wmma execution.
// ---------------------------------------------------------------------------

typedef __attribute__((ext_vector_type(16))) __bf16 v16bf;
typedef __attribute__((ext_vector_type(8)))  float  v8f;

#define BATCH   16
#define CH      128
#define HW      65536
#define SCH     16              // K-split chunks for the Gram phase
#define KCHUNK  (HW / SCH)      // 4096 columns per chunk
#define LDA     40              // padded bf16 row length for 32-wide K tiles
#define LDX     136             // padded bf16 row length for X (128 cols)

union FragU { uint4 u[2]; v16bf v; };

// Pack two fp32 -> one dword of two bf16 (v_cvt_pk_bf16_f32 when available).
#if __has_builtin(__builtin_amdgcn_cvt_pk_bf16_f32)
typedef __attribute__((ext_vector_type(2))) __bf16 v2bf;
__device__ __forceinline__ unsigned pk_bf16(float x, float y) {
    union { v2bf v; unsigned u; } p;
    p.v = __builtin_amdgcn_cvt_pk_bf16_f32(x, y);
    return p.u;
}
#else
__device__ __forceinline__ unsigned pk_bf16(float x, float y) {
    union { __bf16 h[2]; unsigned u; } p;
    p.h[0] = (__bf16)x; p.h[1] = (__bf16)y;
    return p.u;
}
#endif

// 16-bit A-layout fragment (16x32) from an LDS tile:
// lane<16:  M=lane,   K={0..7,16..23}; lane>=16: M=lane-16, K={8..15,24..31}
// -> two contiguous 16-byte ds_load_b128 per lane.
__device__ __forceinline__ v16bf frag_ld(const __bf16* q) {
    FragU f;
    f.u[0] = *(const uint4*)(q);
    f.u[1] = *(const uint4*)(q + 16);
    return f.v;
}

// ---------------------------------------------------------------------------
// Phase 1: partial Gram matrices. grid = (SCH, BATCH), block = 256 (8 waves).
// Each wave owns row-block w. For A·A^T the CDNA5 B-fragment layout is the
// transpose of the A-fragment layout, so the B-frag of column block j is
// bit-identical to the A-frag of row block j: one staged tile feeds both
// operands. (The wave's A-frag is loaded separately from LDS: indexing the
// register array bj[] with runtime w would force a scratch spill.)
// ---------------------------------------------------------------------------
__global__ __launch_bounds__(256) void gram_partials(const float* __restrict__ a,
                                                     float* __restrict__ part) {
    __shared__ __bf16 tile[2][CH * LDA];

    const int b    = blockIdx.y;
    const int s    = blockIdx.x;
    const int t    = threadIdx.x;
    const int lane = t & 31;
    const int w    = t >> 5;
    const int mrow = lane & 15;
    const int k0   = (lane >> 4) * 8;

    const int row  = t >> 1;     // 2 threads per channel row
    const int half = t & 1;      // 16 floats each

    const float* gsrc = a + (size_t)b * CH * HW + (size_t)s * KCHUNK
                          + (size_t)row * HW + half * 16;

    v8f acc[8] = {};

    // ---- prologue: stage K-step 0 into buffer 0 ---------------------------
    {
        const float4* src = (const float4*)gsrc;
        float4 f0 = src[0], f1 = src[1], f2 = src[2], f3 = src[3];
        uint4* dst = (uint4*)&tile[0][row * LDA + half * 16];
        dst[0] = (uint4){ pk_bf16(f0.x, f0.y), pk_bf16(f0.z, f0.w),
                          pk_bf16(f1.x, f1.y), pk_bf16(f1.z, f1.w) };
        dst[1] = (uint4){ pk_bf16(f2.x, f2.y), pk_bf16(f2.z, f2.w),
                          pk_bf16(f3.x, f3.y), pk_bf16(f3.z, f3.w) };
    }
    __syncthreads();

    int p = 0;
    for (int ks = 0; ks < KCHUNK; ks += 32) {
        const bool notlast = (ks + 32 < KCHUNK);

        // ---- issue next-step global loads early (hidden under WMMAs) -----
        float4 g0, g1, g2, g3;
        if (notlast) {
            const float4* src = (const float4*)(gsrc + ks + 32);
            g0 = src[0]; g1 = src[1]; g2 = src[2]; g3 = src[3];
        }

        // ---- gather fragments, then burst 8 WMMAs ------------------------
        const __bf16* tp = tile[p];
        v16bf am = frag_ld(&tp[(w * 16 + mrow) * LDA + k0]);
        v16bf bj[8];
        #pragma unroll
        for (int j = 0; j < 8; ++j)
            bj[j] = frag_ld(&tp[(j * 16 + mrow) * LDA + k0]);
        #pragma unroll
        for (int j = 0; j < 8; ++j)
            acc[j] = __builtin_amdgcn_wmma_f32_16x16x32_bf16(
                false, am, false, bj[j], (short)0, acc[j], false, false);

        // ---- convert + store next tile into the other buffer -------------
        if (notlast) {
            uint4* dst = (uint4*)&tile[p ^ 1][row * LDA + half * 16];
            dst[0] = (uint4){ pk_bf16(g0.x, g0.y), pk_bf16(g0.z, g0.w),
                              pk_bf16(g1.x, g1.y), pk_bf16(g1.z, g1.w) };
            dst[1] = (uint4){ pk_bf16(g2.x, g2.y), pk_bf16(g2.z, g2.w),
                              pk_bf16(g3.x, g3.y), pk_bf16(g3.z, g3.w) };
            __syncthreads();
            p ^= 1;
        }
    }

    // ---- write 128x128 partial (C/D layout: vgpr i, lanes 0-15 -> M=i) ----
    float* pout = part + ((size_t)b * SCH + s) * CH * CH;
    const int mhi = 8 * (lane >> 4);
    #pragma unroll
    for (int j = 0; j < 8; ++j)
        #pragma unroll
        for (int i = 0; i < 8; ++i)
            pout[(size_t)(w * 16 + i + mhi) * CH + j * 16 + (lane & 15)] = acc[j][i];
}

// ---------------------------------------------------------------------------
// Phase 2: deterministic partial reduction + row softmax, store X as bf16.
// grid = (CH, BATCH), block = 128 (one row per block).
// ---------------------------------------------------------------------------
__global__ __launch_bounds__(128) void softmax_rows(const float* __restrict__ part,
                                                    __bf16* __restrict__ xbf) {
    const int b = blockIdx.y, r = blockIdx.x, j = threadIdx.x;

    float v = 0.f;
    for (int s = 0; s < SCH; ++s)
        v += part[(((size_t)b * SCH + s) * CH + r) * CH + j];

    __shared__ float red[CH];
    red[j] = v;
    __syncthreads();
    for (int off = 64; off > 0; off >>= 1) {
        if (j < off) red[j] = fmaxf(red[j], red[j + off]);
        __syncthreads();
    }
    const float m = red[0];
    __syncthreads();

    const float e = __expf(v - m);
    red[j] = e;
    __syncthreads();
    for (int off = 64; off > 0; off >>= 1) {
        if (j < off) red[j] += red[j + off];
        __syncthreads();
    }
    xbf[((size_t)b * CH + r) * CH + j] = (__bf16)(e / red[0]);
}

// ---------------------------------------------------------------------------
// Phase 3: C = X·A, out = beta*C + a. grid = (HW/128, BATCH), block = 256.
// X[b] (bf16, 32KB) resident in LDS; A chunks staged *transposed* and
// double-buffered so column-major B-frag gathers are contiguous b128 DS
// loads and next-chunk global loads overlap the WMMA burst. The epilogue
// re-reads the same 64KB chunk -> L2 hit.
// ---------------------------------------------------------------------------
__global__ __launch_bounds__(256) void attend_kernel(const float* __restrict__ a,
                                                     const __bf16* __restrict__ xbf,
                                                     const float* __restrict__ beta_p,
                                                     float* __restrict__ out) {
    __shared__ __bf16 xs[CH * LDX];       // X[b], row-major, padded
    __shared__ __bf16 ast[2][CH * LDA];   // a' chunk, TRANSPOSED: ast[col][d]

    const int b    = blockIdx.y;
    const int n0   = blockIdx.x * 128;
    const int t    = threadIdx.x;
    const int lane = t & 31;
    const int w    = t >> 5;
    const int mrow = lane & 15;
    const int k0   = (lane >> 4) * 8;

    const float* ab = a + (size_t)b * CH * HW;

    // ---- prologue: X[b] -> LDS, and stage d-block 0 into ast[0] -----------
    const __bf16* xb = xbf + (size_t)b * CH * CH;
    #pragma unroll
    for (int it = 0; it < 8; ++it) {
        int idx8 = it * 256 + t;          // 8-element bf16 group index
        int c  = idx8 >> 4;
        int d8 = idx8 & 15;
        *(uint4*)&xs[c * LDX + d8 * 8] = *(const uint4*)&xb[c * CH + d8 * 8];
    }
    #pragma unroll
    for (int it = 0; it < 4; ++it) {
        int idx = it * 256 + t;           // float4 index within 32x128 chunk
        int dd  = idx >> 5;
        int c4  = (idx & 31) * 4;
        float4 f = *(const float4*)(ab + (size_t)dd * HW + n0 + c4);
        ast[0][(c4 + 0) * LDA + dd] = (__bf16)f.x;
        ast[0][(c4 + 1) * LDA + dd] = (__bf16)f.y;
        ast[0][(c4 + 2) * LDA + dd] = (__bf16)f.z;
        ast[0][(c4 + 3) * LDA + dd] = (__bf16)f.w;
    }
    __syncthreads();

    v8f acc[8] = {};
    int p = 0;
    for (int kk = 0; kk < 4; ++kk) {      // K (=channel d) in steps of 32
        const bool notlast = (kk < 3);

        // ---- next-chunk global loads issued before the WMMA burst --------
        float4 g[4];
        int gdd[4], gc4[4];
        if (notlast) {
            #pragma unroll
            for (int it = 0; it < 4; ++it) {
                int idx = it * 256 + t;
                gdd[it] = idx >> 5;
                gc4[it] = (idx & 31) * 4;
                g[it] = *(const float4*)(ab + (size_t)((kk + 1) * 32 + gdd[it]) * HW
                                            + n0 + gc4[it]);
            }
        }

        // ---- gather fragments, then burst 8 WMMAs ------------------------
        v16bf fb = frag_ld(&ast[p][(w * 16 + mrow) * LDA + k0]);
        v16bf fa[8];
        #pragma unroll
        for (int i = 0; i < 8; ++i)
            fa[i] = frag_ld(&xs[(i * 16 + mrow) * LDX + kk * 32 + k0]);
        #pragma unroll
        for (int i = 0; i < 8; ++i)
            acc[i] = __builtin_amdgcn_wmma_f32_16x16x32_bf16(
                false, fa[i], false, fb, (short)0, acc[i], false, false);

        if (notlast) {
            #pragma unroll
            for (int it = 0; it < 4; ++it) {
                __bf16* d = &ast[p ^ 1][gc4[it] * LDA + gdd[it]];
                d[0 * LDA] = (__bf16)g[it].x;
                d[1 * LDA] = (__bf16)g[it].y;
                d[2 * LDA] = (__bf16)g[it].z;
                d[3 * LDA] = (__bf16)g[it].w;
            }
            __syncthreads();
            p ^= 1;
        }
    }

    // ---- epilogue: out = beta*acc + a (chunk is L2-hot) -------------------
    const float beta = beta_p[0];
    const int col = n0 + w * 16 + (lane & 15);
    const int mhi = 8 * (lane >> 4);
    float* ob = out + (size_t)b * CH * HW;
    #pragma unroll
    for (int i = 0; i < 8; ++i) {
        #pragma unroll
        for (int r = 0; r < 8; ++r) {
            const size_t idx = (size_t)(i * 16 + r + mhi) * HW + col;
            ob[idx] = beta * acc[i][r] + ab[idx];
        }
    }
}

// ---------------------------------------------------------------------------
extern "C" void kernel_launch(void* const* d_in, const int* in_sizes, int n_in,
                              void* d_out, int out_size, void* d_ws, size_t ws_size,
                              hipStream_t stream) {
    (void)in_sizes; (void)n_in; (void)out_size; (void)ws_size;
    const float* a    = (const float*)d_in[0];
    const float* beta = (const float*)d_in[1];
    float*       out  = (float*)d_out;

    // workspace layout: [partials: 16*16*128*128 f32 = 16.8MB][X bf16: 512KB]
    float*  part = (float*)d_ws;
    __bf16* xbf  = (__bf16*)((char*)d_ws + (size_t)BATCH * SCH * CH * CH * sizeof(float));

    gram_partials<<<dim3(SCH, BATCH), 256, 0, stream>>>(a, part);
    softmax_rows <<<dim3(CH, BATCH), 128, 0, stream>>>(part, xbf);
    attend_kernel<<<dim3(HW / 128, BATCH), 256, 0, stream>>>(a, xbf, beta, out);
}
